// xLSTM_83296595739081
// MI455X (gfx1250) — matrix operand order repaired and checked by reading, not compiled, not run on hardware
//
#include <hip/hip_runtime.h>

// ---------------------------------------------------------------------------
// xLSTM forward for MI455X (gfx1250).
// GEMM v3: pure-bf16 operand pipeline. Weights converted to bf16 once per
// launch; activation producers emit bf16 mirrors. GEMM stages bf16 with
// uint4 (b128) global loads and b128 LDS stores -> no conversion VALU in the
// hot loop. 128x64 block tile, 32x32 per wave, 4 WMMAs per 32-wide k-step,
// register double buffering.
// ---------------------------------------------------------------------------

typedef __attribute__((ext_vector_type(16))) __bf16 v16bf;
typedef __attribute__((ext_vector_type(8)))  float  v8f;

constexpr int BATCH = 16;
constexpr int SEQ   = 256;
constexpr int EMB   = 768;
constexpr int HEADS = 4;
constexpr int EIc   = 1536;           // 2*EMB
constexpr int DFFc  = 1024;
constexpr int NACTc = 32;
constexpr int DHM   = EIc / HEADS;    // 384 (mLSTM head dim)
constexpr int DHS   = EMB / HEADS;    // 192 (sLSTM head dim)
constexpr int ROWS  = BATCH * SEQ;    // 4096

#define CDIV(a,b) (((a)+(b)-1)/(b))

// pack two fp32 into packed bf16 dword (RNE)
__device__ __forceinline__ unsigned pk2bf(float a, float b) {
  unsigned ua = __float_as_uint(a);
  unsigned ub = __float_as_uint(b);
  ua += 0x7FFFu + ((ua >> 16) & 1u);
  ub += 0x7FFFu + ((ub >> 16) & 1u);
  return (ua >> 16) | (ub & 0xFFFF0000u);
}
__device__ __forceinline__ __bf16 f2bf1(float f) {
  unsigned u = __float_as_uint(f);
  u += 0x7FFFu + ((u >> 16) & 1u);
  unsigned short h = (unsigned short)(u >> 16);
  __bf16 r;
  __builtin_memcpy(&r, &h, 2);
  return r;
}

__device__ __forceinline__ float siluf(float x) { return x / (1.0f + expf(-x)); }
__device__ __forceinline__ float sigmf(float x) { return 1.0f / (1.0f + expf(-x)); }
__device__ __forceinline__ float geluf(float x) {
  float x3 = x * x * x;
  return 0.5f * x * (1.0f + tanhf(0.7978845608028654f * (x + 0.044715f * x3)));
}

__device__ __forceinline__ float block_sum(float v, float* red) {
  int t = threadIdx.x;
  red[t] = v; __syncthreads();
  for (int o = blockDim.x >> 1; o > 0; o >>= 1) {
    if (t < o) red[t] += red[t + o];
    __syncthreads();
  }
  float r = red[0]; __syncthreads();
  return r;
}
__device__ __forceinline__ float block_max(float v, float* red) {
  int t = threadIdx.x;
  red[t] = v; __syncthreads();
  for (int o = blockDim.x >> 1; o > 0; o >>= 1) {
    if (t < o) red[t] = fmaxf(red[t], red[t + o]);
    __syncthreads();
  }
  float r = red[0]; __syncthreads();
  return r;
}

// fp32 -> bf16 bulk conversion (float4 in, uint2 out), n4 = n/4 groups
__global__ __launch_bounds__(256) void cvt_kernel(const float* __restrict__ in,
    __bf16* __restrict__ out, long long n4) {
  long long i = (long long)blockIdx.x * 256 + threadIdx.x;
  if (i >= n4) return;
  float4 v = *(const float4*)(in + i * 4);
  uint2 u; u.x = pk2bf(v.x, v.y); u.y = pk2bf(v.z, v.w);
  *(uint2*)(out + i * 4) = u;
}

// ---------------------------------------------------------------------------
// Batched strided GEMM: C(f32) = alpha * A(bf16) x B(bf16) (+C).
// Block tile 128x64, 8 wave32 waves in 4(m) x 2(n), each wave 32x32
// (4 v_wmma_f32_16x16x32_bf16 per 32-wide k-step). Requires M%128==0,
// N%64==0, K%32==0 (true for every GEMM in this model).
// Fragment layouts per CDNA5 ISA 7.12.2 (wave32):
//   A 16x32 bf16 : lane L -> row L&15; elems 0..7 -> K=8*(L>>4)+j,
//                  elems 8..15 -> K=16+8*(L>>4)+(j-8)
//   B 32x16 bf16 : lane L -> col L&15; same K order as A
//   C/D 16x16 f32: lane L -> col L&15; elem r -> row r+8*(L>>4)
// LDS rows are K-major, 80B stride (32 bf16 payload + pad): b128-aligned
// stores, 16-row fragment gathers spread across 16 bank groups.
// ---------------------------------------------------------------------------
__global__ __launch_bounds__(256) void gemm_bf16_kernel(
    const __bf16* __restrict__ A, int lda, long long aOffB, long long aOffH,
    const __bf16* __restrict__ Bm, int ldb, long long bOffB, long long bOffH,
    float* __restrict__ C, int ldc, long long cOffB, long long cOffH,
    int K, int batchH, int transB, float alpha, int accum) {
  const int tid  = threadIdx.x;
  const int w    = tid >> 5;
  const int lane = tid & 31;
  const int half = lane >> 4;
  const int l16  = lane & 15;
  const int wm   = w >> 1;       // 0..3 -> 32-row slice
  const int wn   = w & 1;        // 0..1 -> 32-col slice
  const long long m0 = (long long)blockIdx.y * 128;
  const long long n0 = (long long)blockIdx.x * 64;
  const int bz = blockIdx.z;
  const int bb = bz / batchH, hh = bz % batchH;
  A  += (long long)bb * aOffB + (long long)hh * aOffH;
  Bm += (long long)bb * bOffB + (long long)hh * bOffH;
  C  += (long long)bb * cOffB + (long long)hh * cOffH;

  __shared__ unsigned sAu[128][20];   // 40 bf16/row: 32 payload + 8 pad
  __shared__ unsigned sBu[64][20];

  // -------- staging geometry ----------------------------------------------
  // A tile 128x32 bf16 = 512 uint4; 2 per thread. idx8 = tid + i*256.
  const __bf16* aP[2];
  #pragma unroll
  for (int i = 0; i < 2; ++i) {
    int idx8 = tid + i * 256;
    int row = idx8 >> 2, k8 = idx8 & 3;
    aP[i] = A + (m0 + row) * lda + 8 * k8;
  }
  // B tile 64x32 bf16 = 256 uint4.
  const __bf16* bP[2];
  if (transB) {                       // [N,K]: K-contiguous; 1 uint4/thread
    int col = tid >> 2, k8 = tid & 3;
    bP[0] = Bm + (n0 + col) * ldb + 8 * k8;
    bP[1] = bP[0];
  } else {                            // [K,N]: N-contiguous; threads 0..127,
    int kp = tid >> 3;                //  each does k = 2*kp,2*kp+1 x 8 cols
    int n8 = tid & 7;
    const __bf16* base = Bm + (long long)(2 * kp) * ldb + n0 + 8 * n8;
    bP[0] = base;
    bP[1] = base + ldb;
  }

  v8f acc00 = {0.f,0.f,0.f,0.f,0.f,0.f,0.f,0.f};
  v8f acc01 = acc00, acc10 = acc00, acc11 = acc00;

  uint4 ra[2], rb[2];
  // prologue loads (k0 = 0)
  #pragma unroll
  for (int i = 0; i < 2; ++i) ra[i] = *(const uint4*)(aP[i]);
  if (transB) {
    rb[0] = *(const uint4*)(bP[0]);
  } else if (tid < 128) {
    rb[0] = *(const uint4*)(bP[0]);
    rb[1] = *(const uint4*)(bP[1]);
  }

  for (int k0 = 0; k0 < K; k0 += 32) {
    // ---- store staged registers to LDS (raw bf16, no conversion) ----
    #pragma unroll
    for (int i = 0; i < 2; ++i) {
      int idx8 = tid + i * 256;
      int row = idx8 >> 2, k8 = idx8 & 3;
      *(uint4*)&sAu[row][k8 * 4] = ra[i];
    }
    if (transB) {
      int col = tid >> 2, k8 = tid & 3;
      *(uint4*)&sBu[col][k8 * 4] = rb[0];
    } else if (tid < 128) {
      int kp = tid >> 3, n8 = tid & 7;
      const unsigned short* h0 = (const unsigned short*)&rb[0];
      const unsigned short* h1 = (const unsigned short*)&rb[1];
      #pragma unroll
      for (int c = 0; c < 8; ++c)
        sBu[n8 * 8 + c][kp] = (unsigned)h0[c] | ((unsigned)h1[c] << 16);
    }
    __syncthreads();

    // ---- prefetch next k-tile into registers (overlaps the WMMAs) ----
    int kn = k0 + 32;
    if (kn < K) {
      #pragma unroll
      for (int i = 0; i < 2; ++i) ra[i] = *(const uint4*)(aP[i] + kn);
      if (transB) {
        rb[0] = *(const uint4*)(bP[0] + kn);
      } else if (tid < 128) {
        long long koff = (long long)kn * ldb;
        rb[0] = *(const uint4*)(bP[0] + koff);
        rb[1] = *(const uint4*)(bP[1] + koff);
      }
    }

    // ---- build fragments from LDS, issue 4 WMMAs ----
    v16bf aF0, aF1, bF0, bF1;
    {
      const __bf16* ar0 = (const __bf16*)&sAu[wm * 32 + l16][0];
      const __bf16* ar1 = (const __bf16*)&sAu[wm * 32 + 16 + l16][0];
      #pragma unroll
      for (int j = 0; j < 8; ++j) {
        aF0[j]     = ar0[half * 8 + j];
        aF0[8 + j] = ar0[16 + half * 8 + j];
        aF1[j]     = ar1[half * 8 + j];
        aF1[8 + j] = ar1[16 + half * 8 + j];
      }
      const __bf16* br0 = (const __bf16*)&sBu[wn * 32 + l16][0];
      const __bf16* br1 = (const __bf16*)&sBu[wn * 32 + 16 + l16][0];
      #pragma unroll
      for (int j = 0; j < 16; ++j) {
        bF0[j] = br0[half * 8 + j + (j & 8)];
        bF1[j] = br1[half * 8 + j + (j & 8)];
      }
    }
    acc00 = __builtin_amdgcn_wmma_f32_16x16x32_bf16(false, aF0, false, bF0,
                                                    (short)0, acc00, false, false);
    acc01 = __builtin_amdgcn_wmma_f32_16x16x32_bf16(false, aF0, false, bF1,
                                                    (short)0, acc01, false, false);
    acc10 = __builtin_amdgcn_wmma_f32_16x16x32_bf16(false, aF1, false, bF0,
                                                    (short)0, acc10, false, false);
    acc11 = __builtin_amdgcn_wmma_f32_16x16x32_bf16(false, aF1, false, bF1,
                                                    (short)0, acc11, false, false);
    __syncthreads();
  }

  // ---- epilogue: scatter C per ISA C/D layout ----
  const long long mBase = m0 + wm * 32 + 8 * half;
  const long long nBase = n0 + wn * 32 + l16;
  #pragma unroll
  for (int e = 0; e < 8; ++e) {
    long long r0 = (mBase + e) * ldc;
    long long r1 = (mBase + 16 + e) * ldc;
    float v00 = acc00[e] * alpha, v01 = acc01[e] * alpha;
    float v10 = acc10[e] * alpha, v11 = acc11[e] * alpha;
    if (accum) {
      C[r0 + nBase]      += v00;
      C[r0 + nBase + 16] += v01;
      C[r1 + nBase]      += v10;
      C[r1 + nBase + 16] += v11;
    } else {
      C[r0 + nBase]      = v00;
      C[r0 + nBase + 16] = v01;
      C[r1 + nBase]      = v10;
      C[r1 + nBase + 16] = v11;
    }
  }
}

// ---------------------------------------------------------------------------
// LayerNorm over EMB per row; emits f32 + bf16 mirror.
// ---------------------------------------------------------------------------
__global__ __launch_bounds__(256) void ln_kernel(const float* __restrict__ x,
    const float* __restrict__ g, float* __restrict__ out,
    __bf16* __restrict__ outh) {
  __shared__ float red[256];
  long long row = blockIdx.x;
  const float* xr = x + row * EMB;
  float s = 0.f;
  for (int e = threadIdx.x; e < EMB; e += 256) s += xr[e];
  float mean = block_sum(s, red) * (1.0f / EMB);
  float v = 0.f;
  for (int e = threadIdx.x; e < EMB; e += 256) { float d = xr[e] - mean; v += d * d; }
  float var = block_sum(v, red) * (1.0f / EMB);
  float rs = rsqrtf(var + 1e-5f);
  float* orow = out + row * EMB;
  __bf16* hrow = outh + row * EMB;
  for (int e = threadIdx.x; e < EMB; e += 256) {
    float y = (xr[e] - mean) * rs * g[e];
    orow[e] = y;
    hrow[e] = f2bf1(y);
  }
}

// GroupNorm over Dh per (row, head); grid = rows*HEADS, contiguous layout.
__global__ __launch_bounds__(256) void gn_kernel(const float* __restrict__ x,
    const float* __restrict__ g, float* __restrict__ out, int Dh) {
  __shared__ float red[256];
  long long rh = blockIdx.x;
  int h = (int)(rh % HEADS);
  const float* xr = x + rh * (long long)Dh;
  float s = 0.f;
  for (int d = threadIdx.x; d < Dh; d += 256) s += xr[d];
  float mean = block_sum(s, red) / (float)Dh;
  float v = 0.f;
  for (int d = threadIdx.x; d < Dh; d += 256) { float dd = xr[d] - mean; v += dd * dd; }
  float var = block_sum(v, red) / (float)Dh;
  float rs = rsqrtf(var + 1e-5f);
  float* orow = out + rh * (long long)Dh;
  for (int d = threadIdx.x; d < Dh; d += 256) orow[d] = (xr[d] - mean) * rs * g[h * Dh + d];
}

// Depthwise causal conv (K=4) + bias + SiLU; emits f32 + bf16 mirror.
__global__ __launch_bounds__(256) void conv_silu_kernel(const float* __restrict__ x,
    int ldx, const float* __restrict__ w, const float* __restrict__ bias,
    float* __restrict__ out, __bf16* __restrict__ outh, int C) {
  long long idx = (long long)blockIdx.x * 256 + threadIdx.x;
  long long total = (long long)ROWS * C;
  if (idx >= total) return;
  int c = (int)(idx % C);
  long long ts = idx / C;
  int t = (int)(ts % SEQ);
  long long bS = ts - t;
  float acc = bias[c];
  const float* wc = w + (long long)c * 4;
  #pragma unroll
  for (int j = 0; j < 4; ++j) {
    int tt = t + j - 3;
    if (tt >= 0) acc += x[(bS + tt) * ldx + c] * wc[j];
  }
  float y = siluf(acc);
  out[ts * (long long)C + c]  = y;
  outh[ts * (long long)C + c] = f2bf1(y);
}

// 4x4 block-diagonal projection; emits f32 + bf16 mirror.
__global__ __launch_bounds__(256) void blockdiag_kernel(const float* __restrict__ x,
    int ldx, const float* __restrict__ W, float* __restrict__ out,
    __bf16* __restrict__ outh) {
  long long idx = (long long)blockIdx.x * 256 + threadIdx.x;
  if (idx >= (long long)ROWS * EIc) return;
  int c = (int)(idx % EIc);
  long long r = idx / EIc;
  int nb = c >> 2, cc = c & 3;
  const float* xr = x + r * ldx + (nb << 2);
  const float* wp = W + nb * 16 + cc;
  float y = xr[0]*wp[0] + xr[1]*wp[4] + xr[2]*wp[8] + xr[3]*wp[12];
  out[r * (long long)EIc + c]  = y;
  outh[r * (long long)EIc + c] = f2bf1(y);
}

// ig/fg = concat(q,k,v) @ w_{ig,fg} + b (N=4, too narrow for WMMA tiles).
__global__ __launch_bounds__(256) void igfg_kernel(const float* __restrict__ Q,
    const float* __restrict__ Kk, const float* __restrict__ V,
    const float* __restrict__ wig, const float* __restrict__ big,
    const float* __restrict__ wfg, const float* __restrict__ bfg,
    float* __restrict__ IG, float* __restrict__ FG) {
  int r = blockIdx.x * 256 + threadIdx.x;
  if (r >= ROWS) return;
  float ai[4] = {0,0,0,0}, af[4] = {0,0,0,0};
  const float* q = Q  + (long long)r * EIc;
  const float* k = Kk + (long long)r * EIc;
  const float* v = V  + (long long)r * EIc;
  for (int e = 0; e < EIc; ++e) {
    float x = q[e];
    const float* wi = wig + (long long)e * HEADS;
    const float* wf = wfg + (long long)e * HEADS;
    #pragma unroll
    for (int h = 0; h < 4; ++h) { ai[h] += x * wi[h]; af[h] += x * wf[h]; }
  }
  for (int e = 0; e < EIc; ++e) {
    float x = k[e];
    const float* wi = wig + (long long)(EIc + e) * HEADS;
    const float* wf = wfg + (long long)(EIc + e) * HEADS;
    #pragma unroll
    for (int h = 0; h < 4; ++h) { ai[h] += x * wi[h]; af[h] += x * wf[h]; }
  }
  for (int e = 0; e < EIc; ++e) {
    float x = v[e];
    const float* wi = wig + (long long)(2 * EIc + e) * HEADS;
    const float* wf = wfg + (long long)(2 * EIc + e) * HEADS;
    #pragma unroll
    for (int h = 0; h < 4; ++h) { ai[h] += x * wi[h]; af[h] += x * wf[h]; }
  }
  #pragma unroll
  for (int h = 0; h < 4; ++h) {
    IG[(long long)r * 4 + h] = ai[h] + big[h];
    FG[(long long)r * 4 + h] = af[h] + bfg[h];
  }
}

// lc = cumsum_t(log_sigmoid(fg)) per (b,h).
__global__ void cumsum_kernel(const float* __restrict__ FG, float* __restrict__ LC) {
  int idx = threadIdx.x;
  if (idx >= BATCH * HEADS) return;
  int b = idx / HEADS, h = idx % HEADS;
  float run = 0.f;
  for (int t = 0; t < SEQ; ++t) {
    long long i = ((long long)(b * SEQ + t)) * HEADS + h;
    float x = FG[i];
    run += fminf(x, 0.f) - log1pf(expf(-fabsf(x)));
    LC[i] = run;
  }
}

// SC(f32 scores) -> SCh(bf16) = Cm/norm with causal decay matrix.
__global__ __launch_bounds__(256) void decay_kernel(const float* __restrict__ SC,
    __bf16* __restrict__ SCh, const float* __restrict__ LC,
    const float* __restrict__ IG) {
  __shared__ float red[256];
  int id = blockIdx.x;               // (b*HEADS + h)*SEQ + t
  int t  = id % SEQ;
  int bh = id / SEQ;
  int b  = bh / HEADS, h = bh % HEADS;
  int s  = threadIdx.x;              // blockDim == SEQ == 256
  bool valid = (s <= t);
  float lct = LC[((long long)(b * SEQ + t)) * HEADS + h];
  float lcs = LC[((long long)(b * SEQ + s)) * HEADS + h];
  float igs = IG[((long long)(b * SEQ + s)) * HEADS + h];
  float d = valid ? (lct - lcs + igs) : -1e30f;
  float m = block_max(d, red);
  long long si = (((long long)bh) * SEQ + t) * SEQ + s;
  float sc = SC[si];                 // already scaled by Dh^-0.5
  float cm = valid ? expf(d - m) * sc : 0.f;
  float sum = block_sum(cm, red);
  float norm = fmaxf(fabsf(sum), expf(-m)) + 1e-6f;
  SCh[si] = f2bf1(cm / norm);
}

// mLSTM combine: Gh = bf16( (G + skip*xc) * silu(z) ), z = 2nd half of UP.
__global__ __launch_bounds__(256) void combine_kernel(const float* __restrict__ G,
    __bf16* __restrict__ Gh, const float* __restrict__ XC,
    const float* __restrict__ UP, const float* __restrict__ skip) {
  long long idx = (long long)blockIdx.x * 256 + threadIdx.x;
  if (idx >= (long long)ROWS * EIc) return;
  int c = (int)(idx % EIc);
  long long r = idx / EIc;
  float z = UP[r * (long long)(2 * EIc) + EIc + c];
  Gh[idx] = f2bf1((G[idx] + skip[c] * XC[idx]) * siluf(z));
}

// sLSTM recurrence: one block per (b,h), h-state broadcast through LDS.
__global__ __launch_bounds__(192) void scan_kernel(
    const float* __restrict__ IX, const float* __restrict__ FX,
    const float* __restrict__ ZX, const float* __restrict__ OX,
    const float* __restrict__ bi, const float* __restrict__ bf,
    const float* __restrict__ bz, const float* __restrict__ bo,
    const float* __restrict__ Ri, const float* __restrict__ Rf,
    const float* __restrict__ Rz, const float* __restrict__ Ro,
    float* __restrict__ HS) {
  __shared__ float hsh[DHS];
  int bh = blockIdx.x;
  int b = bh / HEADS, h = bh % HEADS;
  int e = threadIdx.x;
  const float* ri_w = Ri + (long long)h * DHS * DHS;
  const float* rf_w = Rf + (long long)h * DHS * DHS;
  const float* rz_w = Rz + (long long)h * DHS * DHS;
  const float* ro_w = Ro + (long long)h * DHS * DHS;
  float bie = bi[h * DHS + e], bfe = bf[h * DHS + e];
  float bze = bz[h * DHS + e], boe = bo[h * DHS + e];
  float c = 0.f, n = 0.f, m = -30.f;
  hsh[e] = 0.f;
  __syncthreads();
  for (int t = 0; t < SEQ; ++t) {
    long long base = ((long long)(b * SEQ + t)) * EMB + h * DHS;
    float ri = IX[base + e] + bie;
    float rf = FX[base + e] + bfe;
    float rz = ZX[base + e] + bze;
    float ro = OX[base + e] + boe;
    for (int d = 0; d < DHS; ++d) {
      float hv = hsh[d];
      int wi = d * DHS + e;
      ri += hv * ri_w[wi];
      rf += hv * rf_w[wi];
      rz += hv * rz_w[wi];
      ro += hv * ro_w[wi];
    }
    float mn  = fmaxf(rf + m, ri);
    float igt = expf(ri - mn);
    float fgt = expf(rf + m - mn);
    c = fgt * c + igt * tanhf(rz);
    n = fgt * n + igt;
    float hnew = sigmf(ro) * c / (n + 1e-6f);
    m = mn;
    __syncthreads();
    hsh[e] = hnew;
    HS[base + e] = hnew;
    __syncthreads();
  }
}

__global__ __launch_bounds__(256) void add_kernel(float* __restrict__ X,
    const float* __restrict__ A) {
  long long idx = (long long)blockIdx.x * 256 + threadIdx.x;
  if (idx >= (long long)ROWS * EMB) return;
  X[idx] += A[idx];
}

// slstm FF gate: XCh = bf16( gelu(a) * g )
__global__ __launch_bounds__(256) void gelugate_kernel(const float* __restrict__ UP,
    __bf16* __restrict__ Oh) {
  long long idx = (long long)blockIdx.x * 256 + threadIdx.x;
  if (idx >= (long long)ROWS * DFFc) return;
  int d = (int)(idx % DFFc);
  long long r = idx / DFFc;
  float a = UP[r * (long long)(2 * DFFc) + d];
  float g = UP[r * (long long)(2 * DFFc) + DFFc + d];
  Oh[r * (long long)DFFc + d] = f2bf1(geluf(a) * g);
}

__global__ __launch_bounds__(256) void head_kernel(const float* __restrict__ XN,
    const float* __restrict__ wt, const float* __restrict__ bt,
    const float* __restrict__ wp, const float* __restrict__ bp,
    float* __restrict__ outp) {
  int idx = blockIdx.x * 256 + threadIdx.x;
  if (idx < BATCH * NACTc) {
    int b = idx / NACTc, j = idx % NACTc;
    const float* o = XN + ((long long)(b * SEQ + SEQ - 1)) * EMB;
    float acc = bt[j];
    for (int e = 0; e < EMB; ++e) acc += o[e] * wt[e * NACTc + j];
    outp[idx] = acc;
  } else if (idx < BATCH * NACTc + BATCH * 3) {
    int k = idx - BATCH * NACTc;
    int b = k / 3, j = k % 3;
    const float* o = XN + ((long long)(b * SEQ + SEQ - 1)) * EMB;
    float acc = bp[j];
    for (int e = 0; e < EMB; ++e) acc += o[e] * wp[e * 3 + j];
    outp[BATCH * NACTc + k] = acc;
  }
}

// ---------------------------------------------------------------------------
// Host side
// ---------------------------------------------------------------------------
struct MP {
  const float *b_fg, *b_ig, *conv_b, *conv_w, *gn, *ln, *skip, *w_down,
              *w_fg, *w_ig, *w_up, *wk, *wq, *wv;
};
struct SP {
  const float *bf, *bi, *bo, *bz, *conv_b, *conv_w, *gn, *ln, *ln2,
              *rf, *ri, *ro, *rz, *w_ff_down, *w_ff_up, *wf, *wi, *wo, *wz;
};
struct HP { const float *b_p, *b_t, *ln_out, *w_p, *w_t; };

struct Bufs {
  float *X, *XN, *UP, *XC, *Qb, *Kb, *Vb, *SCb, *Hb, *Gb, *IGb, *FGb, *LCb;
  __bf16 *XNh, *XCh, *Qh, *Kh, *Vh, *SCh, *Gh;
};

static inline void launch_gemm(hipStream_t st,
    const __bf16* A, int lda, long long aOffB, long long aOffH,
    const __bf16* Bm, int ldb, long long bOffB, long long bOffH,
    float* C, int ldc, long long cOffB, long long cOffH,
    int M, int N, int K, int batchB, int batchH, int transB,
    float alpha, int accum) {
  dim3 g(N / 64, M / 128, batchB * batchH);
  gemm_bf16_kernel<<<g, 256, 0, st>>>(A, lda, aOffB, aOffH, Bm, ldb, bOffB, bOffH,
                                      C, ldc, cOffB, cOffH, K, batchH, transB,
                                      alpha, accum);
}

static inline void cvt(hipStream_t st, const float* in, __bf16* out, long long n) {
  long long n4 = n / 4;
  cvt_kernel<<<(unsigned)CDIV(n4, 256), 256, 0, st>>>(in, out, n4);
}

static void run_mlstm(hipStream_t st, const MP& p, const Bufs& B,
                      const __bf16* wUpH, const __bf16* wDownH) {
  ln_kernel<<<ROWS, 256, 0, st>>>(B.X, p.ln, B.XN, B.XNh);
  launch_gemm(st, B.XNh, EMB, 0, 0, wUpH, 2 * EIc, 0, 0, B.UP, 2 * EIc, 0, 0,
              ROWS, 2 * EIc, EMB, 1, 1, 0, 1.0f, 0);
  conv_silu_kernel<<<CDIV(ROWS * EIc, 256), 256, 0, st>>>(B.UP, 2 * EIc, p.conv_w,
                                                          p.conv_b, B.XC, B.XCh, EIc);
  blockdiag_kernel<<<CDIV(ROWS * EIc, 256), 256, 0, st>>>(B.XC, EIc, p.wq, B.Qb, B.Qh);
  blockdiag_kernel<<<CDIV(ROWS * EIc, 256), 256, 0, st>>>(B.XC, EIc, p.wk, B.Kb, B.Kh);
  blockdiag_kernel<<<CDIV(ROWS * EIc, 256), 256, 0, st>>>(B.UP, 2 * EIc, p.wv, B.Vb, B.Vh);
  igfg_kernel<<<CDIV(ROWS, 256), 256, 0, st>>>(B.Qb, B.Kb, B.Vb, p.w_ig, p.b_ig,
                                               p.w_fg, p.b_fg, B.IGb, B.FGb);
  cumsum_kernel<<<1, 64, 0, st>>>(B.FGb, B.LCb);
  // scores[b,h] = (q[b,:,h,:] @ k[b,:,h,:]^T) * Dh^-0.5
  launch_gemm(st, B.Qh, EIc, (long long)SEQ * EIc, DHM,
              B.Kh, EIc, (long long)SEQ * EIc, DHM,
              B.SCb, SEQ, (long long)HEADS * SEQ * SEQ, (long long)SEQ * SEQ,
              SEQ, SEQ, DHM, BATCH, HEADS, 1, 0.05103103630798288f, 0);
  decay_kernel<<<BATCH * HEADS * SEQ, 256, 0, st>>>(B.SCb, B.SCh, B.LCb, B.IGb);
  // h = (Cm/norm) @ v
  launch_gemm(st, B.SCh, SEQ, (long long)HEADS * SEQ * SEQ, (long long)SEQ * SEQ,
              B.Vh, EIc, (long long)SEQ * EIc, DHM,
              B.Hb, EIc, (long long)SEQ * EIc, DHM,
              SEQ, DHM, SEQ, BATCH, HEADS, 0, 1.0f, 0);
  gn_kernel<<<ROWS * HEADS, 256, 0, st>>>(B.Hb, p.gn, B.Gb, DHM);
  combine_kernel<<<CDIV(ROWS * EIc, 256), 256, 0, st>>>(B.Gb, B.Gh, B.XC, B.UP, p.skip);
  launch_gemm(st, B.Gh, EIc, 0, 0, wDownH, EMB, 0, 0, B.X, EMB, 0, 0,
              ROWS, EMB, EIc, 1, 1, 0, 1.0f, 1);   // residual accumulate
}

static void run_slstm(hipStream_t st, const SP& p, const Bufs& B,
                      const __bf16* wiH, const __bf16* wfH, const __bf16* wzH,
                      const __bf16* woH, const __bf16* ffUpH, const __bf16* ffDownH) {
  ln_kernel<<<ROWS, 256, 0, st>>>(B.X, p.ln, B.XN, B.XNh);
  conv_silu_kernel<<<CDIV(ROWS * EMB, 256), 256, 0, st>>>(B.XN, EMB, p.conv_w,
                                                          p.conv_b, B.XC, B.XCh, EMB);
  // per-head gate projections (batched over HEADS via strided views)
  launch_gemm(st, B.XCh, EMB, 0, DHS, wiH, DHS, 0, (long long)DHS * DHS,
              B.Qb, EMB, 0, DHS, ROWS, DHS, DHS, 1, HEADS, 0, 1.0f, 0);
  launch_gemm(st, B.XCh, EMB, 0, DHS, wfH, DHS, 0, (long long)DHS * DHS,
              B.Kb, EMB, 0, DHS, ROWS, DHS, DHS, 1, HEADS, 0, 1.0f, 0);
  launch_gemm(st, B.XNh, EMB, 0, DHS, wzH, DHS, 0, (long long)DHS * DHS,
              B.Vb, EMB, 0, DHS, ROWS, DHS, DHS, 1, HEADS, 0, 1.0f, 0);
  launch_gemm(st, B.XNh, EMB, 0, DHS, woH, DHS, 0, (long long)DHS * DHS,
              B.Hb, EMB, 0, DHS, ROWS, DHS, DHS, 1, HEADS, 0, 1.0f, 0);
  scan_kernel<<<BATCH * HEADS, DHS, 0, st>>>(B.Qb, B.Kb, B.Vb, B.Hb,
                                             p.bi, p.bf, p.bz, p.bo,
                                             p.ri, p.rf, p.rz, p.ro, B.Gb);
  gn_kernel<<<ROWS * HEADS, 256, 0, st>>>(B.Gb, p.gn, B.SCb, DHS);
  add_kernel<<<CDIV(ROWS * EMB, 256), 256, 0, st>>>(B.X, B.SCb);
  ln_kernel<<<ROWS, 256, 0, st>>>(B.X, p.ln2, B.XN, B.XNh);
  launch_gemm(st, B.XNh, EMB, 0, 0, ffUpH, 2 * DFFc, 0, 0, B.UP, 2 * DFFc, 0, 0,
              ROWS, 2 * DFFc, EMB, 1, 1, 0, 1.0f, 0);
  gelugate_kernel<<<CDIV(ROWS * DFFc, 256), 256, 0, st>>>(B.UP, B.XCh);
  launch_gemm(st, B.XCh, DFFc, 0, 0, ffDownH, EMB, 0, 0, B.X, EMB, 0, 0,
              ROWS, EMB, DFFc, 1, 1, 0, 1.0f, 1);  // residual accumulate
}

extern "C" void kernel_launch(void* const* d_in, const int* in_sizes, int n_in,
                              void* d_out, int out_size, void* d_ws, size_t ws_size,
                              hipStream_t stream) {
  (void)in_sizes; (void)n_in; (void)out_size; (void)ws_size;
  int pi = 0;
  auto nxt = [&]() -> const float* { return (const float*)d_in[pi++]; };
  const float* ctx = nxt();

  // params flattened in pytree (sorted-key) order: blocks[0..3] then head
  MP mp[3]; SP sp; HP hp;
  int mi = 0;
  for (int b = 0; b < 4; ++b) {
    if (b == 1) {  // sLSTM block
      sp.bf = nxt(); sp.bi = nxt(); sp.bo = nxt(); sp.bz = nxt();
      sp.conv_b = nxt(); sp.conv_w = nxt(); sp.gn = nxt();
      sp.ln = nxt(); sp.ln2 = nxt();
      sp.rf = nxt(); sp.ri = nxt(); sp.ro = nxt(); sp.rz = nxt();
      sp.w_ff_down = nxt(); sp.w_ff_up = nxt();
      sp.wf = nxt(); sp.wi = nxt(); sp.wo = nxt(); sp.wz = nxt();
    } else {       // mLSTM block
      MP& p = mp[mi++];
      p.b_fg = nxt(); p.b_ig = nxt(); p.conv_b = nxt(); p.conv_w = nxt();
      p.gn = nxt(); p.ln = nxt(); p.skip = nxt(); p.w_down = nxt();
      p.w_fg = nxt(); p.w_ig = nxt(); p.w_up = nxt();
      p.wk = nxt(); p.wq = nxt(); p.wv = nxt();
    }
  }
  hp.b_p = nxt(); hp.b_t = nxt(); hp.ln_out = nxt(); hp.w_p = nxt(); hp.w_t = nxt();

  float* ws = (float*)d_ws;
  size_t off = 0;
  auto alloc = [&](size_t n) { float* p = ws + off; off += n; return p; };
  Bufs B;
  B.X   = alloc((size_t)ROWS * EMB);
  B.XN  = alloc((size_t)ROWS * EMB);
  B.UP  = alloc((size_t)ROWS * 2 * EIc);
  B.XC  = alloc((size_t)ROWS * EIc);
  B.Qb  = alloc((size_t)ROWS * EIc);
  B.Kb  = alloc((size_t)ROWS * EIc);
  B.Vb  = alloc((size_t)ROWS * EIc);
  B.SCb = alloc((size_t)BATCH * HEADS * SEQ * SEQ);
  B.Hb  = alloc((size_t)ROWS * EIc);
  B.Gb  = alloc((size_t)ROWS * EIc);
  B.IGb = alloc((size_t)ROWS * HEADS);
  B.FGb = alloc((size_t)ROWS * HEADS);
  B.LCb = alloc((size_t)ROWS * HEADS);

  // bf16 arena (placed after the f32 arena; all counts are multiples of 8)
  __bf16* wsH = (__bf16*)(ws + off);
  size_t offH = 0;
  auto allocH = [&](size_t n) { __bf16* p = wsH + offH; offH += n; return p; };
  B.XNh = allocH((size_t)ROWS * EMB);
  B.XCh = allocH((size_t)ROWS * EIc);
  B.Qh  = allocH((size_t)ROWS * EIc);
  B.Kh  = allocH((size_t)ROWS * EIc);
  B.Vh  = allocH((size_t)ROWS * EIc);
  B.SCh = allocH((size_t)BATCH * HEADS * SEQ * SEQ);
  B.Gh  = allocH((size_t)ROWS * EIc);
  __bf16* wUpH[3];
  __bf16* wDownH[3];
  for (int i = 0; i < 3; ++i) {
    wUpH[i]   = allocH((size_t)EMB * 2 * EIc);
    wDownH[i] = allocH((size_t)EIc * EMB);
  }
  __bf16* wiH = allocH((size_t)HEADS * DHS * DHS);
  __bf16* wfH = allocH((size_t)HEADS * DHS * DHS);
  __bf16* wzH = allocH((size_t)HEADS * DHS * DHS);
  __bf16* woH = allocH((size_t)HEADS * DHS * DHS);
  __bf16* ffUpH   = allocH((size_t)EMB * 2 * DFFc);
  __bf16* ffDownH = allocH((size_t)DFFc * EMB);

  // one-time (per launch) weight conversions to bf16
  for (int i = 0; i < 3; ++i) {
    cvt(stream, mp[i].w_up,   wUpH[i],   (long long)EMB * 2 * EIc);
    cvt(stream, mp[i].w_down, wDownH[i], (long long)EIc * EMB);
  }
  cvt(stream, sp.wi, wiH, (long long)HEADS * DHS * DHS);
  cvt(stream, sp.wf, wfH, (long long)HEADS * DHS * DHS);
  cvt(stream, sp.wz, wzH, (long long)HEADS * DHS * DHS);
  cvt(stream, sp.wo, woH, (long long)HEADS * DHS * DHS);
  cvt(stream, sp.w_ff_up,   ffUpH,   (long long)EMB * 2 * DFFc);
  cvt(stream, sp.w_ff_down, ffDownH, (long long)DFFc * EMB);

  hipMemcpyAsync(B.X, ctx, (size_t)ROWS * EMB * sizeof(float),
                 hipMemcpyDeviceToDevice, stream);

  for (int b = 0; b < 4; ++b) {
    if (b == 1) run_slstm(stream, sp, B, wiH, wfH, wzH, woH, ffUpH, ffDownH);
    else {
      int i = (b == 0) ? 0 : b - 1;
      run_mlstm(stream, mp[i], B, wUpH[i], wDownH[i]);
    }
  }

  ln_kernel<<<ROWS, 256, 0, stream>>>(B.X, hp.ln_out, B.XN, B.XNh);
  head_kernel<<<3, 256, 0, stream>>>(B.XN, hp.w_t, hp.b_t, hp.w_p, hp.b_p,
                                     (float*)d_out);
}
